// GradMimicryNetwork_44710609551491
// MI455X (gfx1250) — matrix operand order repaired and checked by reading, not compiled, not run on hardware
//
#include <hip/hip_runtime.h>

// ---------------------------------------------------------------------------
// LIF SNN forward, restructured layer-by-layer:
//   GEMM (T*B x K x N, f16 WMMA, f32 accum)  ->  elementwise LIF scan over t
// The t-1 spike dependency becomes a 128-row (one timestep) zero-padded shift
// of the GEMM A-operand. All matrix math runs on v_wmma_f32_16x16x32_f16.
// GEMM is templated on (N, K), double-buffers LDS, prefetches the K+64 tile
// (global_prefetch_b8), and keeps all staging data in named registers
// (bit_cast, no address-taken locals -> no scratch spills).
// ---------------------------------------------------------------------------

typedef _Float16 f16;
typedef __attribute__((ext_vector_type(16))) _Float16 v16h;
typedef __attribute__((ext_vector_type(8)))  _Float16 v8h;
typedef __attribute__((ext_vector_type(8)))  float    v8f;

#define TSTEPS 256
#define BATCH  128
#define MROWS  (TSTEPS * BATCH)   // 32768 GEMM rows

struct u4x2 { uint4 lo, hi; };

// assemble a 16-half WMMA fragment from two 16B LDS chunks, register-clean
static __device__ __forceinline__ v16h ld_frag(const f16* p0, const f16* p1) {
  u4x2 t;
  t.lo = *(const uint4*)p0;
  t.hi = *(const uint4*)p1;
  return __builtin_bit_cast(v16h, t);
}

// ---------------------------------------------------------------- weights ---
__global__ void relu_to_f16(const float* __restrict__ w, f16* __restrict__ o, int n) {
  int i = blockIdx.x * 256 + threadIdx.x;
  if (i < n) {
    float x = w[i];
    o[i] = (f16)(x > 0.f ? x : 0.f);
  }
}

__global__ void zero_f16_kern(f16* __restrict__ p, int n) {
  int i = blockIdx.x * 256 + threadIdx.x;
  if (i < n) p[i] = (f16)0.f;
}

// ------------------------------------------------- input (B,I,T) -> (T*B,I) --
__global__ __launch_bounds__(256) void transpose_in(const float* __restrict__ in,
                                                    f16* __restrict__ a0) {
  __shared__ f16 tile[32][33];
  const int b  = blockIdx.z;
  const int t0 = blockIdx.x * 32;
  const int i0 = blockIdx.y * 32;
  const int tx = threadIdx.x, ty = threadIdx.y;
  const float* src = in + ((size_t)b * 1024 + i0) * 256 + t0;
#pragma unroll
  for (int k = 0; k < 4; ++k) {
    int ii = ty + 8 * k;
    tile[ii][tx] = (f16)src[(size_t)ii * 256 + tx];
  }
  __syncthreads();
#pragma unroll
  for (int k = 0; k < 4; ++k) {
    int tt = ty + 8 * k;
    a0[((size_t)(t0 + tt) * BATCH + b) * 1024 + i0 + tx] = tile[tx][tt];
  }
}

// ----------------------------------------------------------------- GEMM -----
// C[M=32768,N] = A[M,K] * B[K,N], f16 in / f32 out. N, K compile-time.
// 256 threads = 8 wave32; macro tile 128x128; K-step 32; LDS double-buffered.
// Wave grid 2(M) x 4(N); each wave owns 64x32 of C = 4x2 16x16 fragments.
template <int N, int K>
__global__ __launch_bounds__(256) void gemm_wmma(const f16* __restrict__ A,
                                                 const f16* __restrict__ B,
                                                 float* __restrict__ C) {
  constexpr int PAD = 40;  // padded row length in halves (80B, 16B-aligned)
  __shared__ __align__(16) f16 As[2][128 * PAD];
  __shared__ __align__(16) f16 Bs[2][128 * PAD];

  const int tid   = threadIdx.x;
  const int lane  = tid & 31;
  const int w     = tid >> 5;
  const int waveM = w >> 2;        // 0..1
  const int waveN = w & 3;         // 0..3
  const int rowBase = blockIdx.y * 128;
  const int colBase = blockIdx.x * 128;
  const int hi  = (lane >> 4) & 1; // lane half (ISA 7.12.2 layout split)
  const int l16 = lane & 15;

  // staging coordinates (512 16B segments per tile, 2 per thread)
  const int arow0 = tid >> 2;              // A: rows 0..63 (seg0), +64 (seg1)
  const int ac8   = (tid & 3) * 8;
  const int bk0   = tid >> 4;              // B: k 0..15 (seg0), +16 (seg1)
  const int bc8   = (tid & 15) * 8;

  // global staging pointers (k0 added per step)
  const f16* Ag  = A + (size_t)rowBase * K;
  const f16* Bg  = B + colBase;
  const f16* Ap0 = Ag + (size_t)arow0 * K + ac8;
  const f16* Ap1 = Ag + (size_t)(arow0 + 64) * K + ac8;
  const f16* Bp0 = Bg + (size_t)bk0 * N + bc8;
  const f16* Bp1 = Bg + (size_t)(bk0 + 16) * N + bc8;

  v8f acc[4][2];
#pragma unroll
  for (int mf = 0; mf < 4; ++mf)
#pragma unroll
    for (int nf = 0; nf < 2; ++nf)
      acc[mf][nf] = (v8f){0.f, 0.f, 0.f, 0.f, 0.f, 0.f, 0.f, 0.f};

  uint4 ra0, ra1, rb0, rb1;  // named staging regs: never address-taken

#define LOAD_TILES(k0)                                              \
  do {                                                              \
    ra0 = *(const uint4*)(Ap0 + (k0));                              \
    ra1 = *(const uint4*)(Ap1 + (k0));                              \
    rb0 = *(const uint4*)(Bp0 + (size_t)(k0)*N);                    \
    rb1 = *(const uint4*)(Bp1 + (size_t)(k0)*N);                    \
  } while (0)

#define STORE_TILES(p)                                              \
  do {                                                              \
    *(uint4*)(&As[p][arow0 * PAD + ac8]) = ra0;                     \
    *(uint4*)(&As[p][(arow0 + 64) * PAD + ac8]) = ra1;              \
    v8h hb0 = __builtin_bit_cast(v8h, rb0);                         \
    v8h hb1 = __builtin_bit_cast(v8h, rb1);                         \
    _Pragma("unroll")                                               \
    for (int j = 0; j < 8; ++j) {                                   \
      Bs[p][(bc8 + j) * PAD + bk0]      = hb0[j];                   \
      Bs[p][(bc8 + j) * PAD + bk0 + 16] = hb1[j];                   \
    }                                                               \
  } while (0)

  LOAD_TILES(0);
  STORE_TILES(0);
  __syncthreads();

  int p = 0;
#pragma unroll 1
  for (int k0 = 0; k0 < K; k0 += 32) {
    const bool hasNext = (k0 + 32 < K);
    if (hasNext) LOAD_TILES(k0 + 32);             // overlaps with WMMA below
    if (k0 + 64 < K) {                            // gfx1250 global_prefetch_b8
      __builtin_prefetch(Ap0 + (k0 + 64), 0, 3);
      __builtin_prefetch(Bp0 + (size_t)(k0 + 64) * N, 0, 3);
    }

    // ---- all 6 fragments up front: one dscnt wait, then 8 back-to-back WMMAs
    const f16* bpBase = &Bs[p][0];
    const f16* apBase = &As[p][0];
    const f16* bp0 = bpBase + (waveN * 32 + l16) * PAD + hi * 8;
    const f16* bp1 = bp0 + 16 * PAD;
    v16h bf0 = ld_frag(bp0, bp0 + 16);
    v16h bf1 = ld_frag(bp1, bp1 + 16);
    const f16* ap = apBase + (waveM * 64 + l16) * PAD + hi * 8;
    v16h af0 = ld_frag(ap,            ap + 16);
    v16h af1 = ld_frag(ap + 16 * PAD, ap + 16 * PAD + 16);
    v16h af2 = ld_frag(ap + 32 * PAD, ap + 32 * PAD + 16);
    v16h af3 = ld_frag(ap + 48 * PAD, ap + 48 * PAD + 16);

    acc[0][0] = __builtin_amdgcn_wmma_f32_16x16x32_f16(false, af0, false, bf0, (short)0, acc[0][0], false, false);
    acc[0][1] = __builtin_amdgcn_wmma_f32_16x16x32_f16(false, af0, false, bf1, (short)0, acc[0][1], false, false);
    acc[1][0] = __builtin_amdgcn_wmma_f32_16x16x32_f16(false, af1, false, bf0, (short)0, acc[1][0], false, false);
    acc[1][1] = __builtin_amdgcn_wmma_f32_16x16x32_f16(false, af1, false, bf1, (short)0, acc[1][1], false, false);
    acc[2][0] = __builtin_amdgcn_wmma_f32_16x16x32_f16(false, af2, false, bf0, (short)0, acc[2][0], false, false);
    acc[2][1] = __builtin_amdgcn_wmma_f32_16x16x32_f16(false, af2, false, bf1, (short)0, acc[2][1], false, false);
    acc[3][0] = __builtin_amdgcn_wmma_f32_16x16x32_f16(false, af3, false, bf0, (short)0, acc[3][0], false, false);
    acc[3][1] = __builtin_amdgcn_wmma_f32_16x16x32_f16(false, af3, false, bf1, (short)0, acc[3][1], false, false);

    if (hasNext) {
      STORE_TILES(p ^ 1);  // disjoint from buffer p being read
      __syncthreads();
      p ^= 1;
    }
  }
#undef LOAD_TILES
#undef STORE_TILES

  // ---- epilogue: C element (i of v8f) -> M = i + hi*8, N = l16.
  // N is compile-time, so row strides fold into immediate store offsets.
  float* cp = C + (size_t)(rowBase + waveM * 64 + hi * 8) * N
                + (colBase + waveN * 32 + l16);
#pragma unroll
  for (int mf = 0; mf < 4; ++mf)
#pragma unroll
    for (int nf = 0; nf < 2; ++nf)
#pragma unroll
      for (int i = 0; i < 8; ++i)
        cp[(size_t)(mf * 16 + i) * N + nf * 16] = acc[mf][nf][i];
}

// ------------------------------------------------------------- LIF scan -----
__global__ __launch_bounds__(256) void lif_scan(const float* __restrict__ cur,
                                                float* __restrict__ spk_out,
                                                f16* __restrict__ s16,
                                                float* __restrict__ rates,
                                                int H) {
  int tid = blockIdx.x * 256 + threadIdx.x;
  if (tid >= BATCH * H) return;
  int b = tid / H, h = tid % H;

  const size_t stride = (size_t)BATCH * H;      // one timestep of (t,b,h)
  const float* cp = cur + (size_t)b * H + h;
  f16* sp16 = s16 ? s16 + stride + (size_t)b * H + h : nullptr;
  float* outp = spk_out + ((size_t)b * H + h) * TSTEPS;

  float v = -65.f;     // V_REST
  float cnt = 0.f;

#pragma unroll 1
  for (int t = 0; t < TSTEPS; t += 4) {
    float ss[4];
#pragma unroll
    for (int j = 0; j < 4; ++j) {
      float c  = *cp;  cp += stride;
      float vn = v + 0.05f * ((-65.f - v) + c);          // DT/TAU = 1/20
      float s  = (vn >= -50.f) ? 1.f : 0.f;              // V_TH
      v = (s > 0.f) ? -65.f : vn;                        // reset to V_RESET
      ss[j] = s;
      cnt += s;
      if (sp16) { *sp16 = (f16)s; sp16 += stride; }
    }
    *(float4*)(outp + t) = make_float4(ss[0], ss[1], ss[2], ss[3]);
  }
  if (rates) rates[tid] = cnt * (1.f / 256.f);
}

// --------------------------------------------------------------- driver -----
extern "C" void kernel_launch(void* const* d_in, const int* in_sizes, int n_in,
                              void* d_out, int out_size, void* d_ws, size_t ws_size,
                              hipStream_t stream) {
  (void)in_sizes; (void)n_in; (void)out_size;
  const float* w[5];
  for (int i = 0; i < 5; ++i) w[i] = (const float*)d_in[i];
  const float* in_spk = (const float*)d_in[5];
  float* out = (float*)d_out;

  // workspace carving (256B aligned)
  char* ws = (char*)d_ws;
  size_t off = 0;
  auto carve = [&](size_t bytes) -> char* {
    off = (off + 255) & ~(size_t)255;
    char* p = ws + off;
    off += bytes;
    return p;
  };
  f16* rw0 = (f16*)carve((size_t)1024 * 1024 * 2);
  f16* rw1 = (f16*)carve((size_t)1024 * 1024 * 2);
  f16* rw2 = (f16*)carve((size_t)1024 * 512 * 2);
  f16* rw3 = (f16*)carve((size_t)512 * 256 * 2);
  f16* rw4 = (f16*)carve((size_t)256 * 128 * 2);
  float* cur = (float*)carve((size_t)MROWS * 1024 * 4);
  // spike ping-pong buffers: (T+1)*B rows, 128 leading zero rows = t-1 shift.
  // sA doubles as the transposed-input buffer (dead after GEMM0).
  f16* sA = (f16*)carve((size_t)(MROWS + BATCH) * 1024 * 2);
  f16* sB = (f16*)carve((size_t)(MROWS + BATCH) * 1024 * 2);
  if (off > ws_size) return;  // workspace too small; nothing sane to do
  f16* a0 = sA;

  // output layout (flat f32): h0, h1, h2, h3, out_spikes, rates
  float* o_h0 = out;
  float* o_h1 = out + (size_t)33554432;
  float* o_h2 = out + (size_t)67108864;
  float* o_h3 = out + (size_t)83886080;
  float* o_os = out + (size_t)92274688;
  float* o_fr = out + (size_t)96468992;

  // 1) relu+cast weights
  const int wn[5] = {1048576, 1048576, 524288, 131072, 32768};
  f16* rw[5] = {rw0, rw1, rw2, rw3, rw4};
  for (int i = 0; i < 5; ++i)
    relu_to_f16<<<(wn[i] + 255) / 256, 256, 0, stream>>>(w[i], rw[i], wn[i]);

  // 2) transpose input spikes -> a0 (f16, GEMM A layout)
  transpose_in<<<dim3(TSTEPS / 32, 1024 / 32, BATCH), dim3(32, 8), 0, stream>>>(in_spk, a0);

  // 3) layer 0 input current: cur = a0 @ relu(w0)
  gemm_wmma<1024, 1024><<<dim3(8, MROWS / 128), 256, 0, stream>>>(a0, rw0, cur);

  // a0 is dead now: re-zero the t=-1 rows of both spike buffers (every launch,
  // so graph replay is deterministic).
  zero_f16_kern<<<(BATCH * 1024 + 255) / 256, 256, 0, stream>>>(sA, BATCH * 1024);
  zero_f16_kern<<<(BATCH * 1024 + 255) / 256, 256, 0, stream>>>(sB, BATCH * 1024);

  // 4) LIF layer 0 -> spikes s0 (f32 out + shifted f16 for next GEMM)
  lif_scan<<<(BATCH * 1024 + 255) / 256, 256, 0, stream>>>(cur, o_h0, sA, nullptr, 1024);

  // 5) layer 1: cur = s0[t-1] @ relu(w1)
  gemm_wmma<1024, 1024><<<dim3(8, MROWS / 128), 256, 0, stream>>>(sA, rw1, cur);
  lif_scan<<<(BATCH * 1024 + 255) / 256, 256, 0, stream>>>(cur, o_h1, sB, nullptr, 1024);

  // 6) layer 2 (H=512)
  gemm_wmma<512, 1024><<<dim3(4, MROWS / 128), 256, 0, stream>>>(sB, rw2, cur);
  lif_scan<<<(BATCH * 512 + 255) / 256, 256, 0, stream>>>(cur, o_h2, sA, nullptr, 512);

  // 7) layer 3 (H=256)
  gemm_wmma<256, 512><<<dim3(2, MROWS / 128), 256, 0, stream>>>(sA, rw3, cur);
  lif_scan<<<(BATCH * 256 + 255) / 256, 256, 0, stream>>>(cur, o_h3, sB, nullptr, 256);

  // 8) output layer (H=128) + firing rates
  gemm_wmma<128, 256><<<dim3(1, MROWS / 128), 256, 0, stream>>>(sB, rw4, cur);
  lif_scan<<<(BATCH * 128 + 255) / 256, 256, 0, stream>>>(cur, o_os, nullptr, o_fr, 128);
}
